// BSplineACTF_14920716386768
// MI455X (gfx1250) — matrix operand order repaired and checked by reading, not compiled, not run on hardware
//
#include <hip/hip_runtime.h>
#include <hip/hip_bf16.h>
#include <stdint.h>

// ---------------------------------------------------------------------------
// KAN B-spline layer as fused (spline-expand -> f16 WMMA GEMM) for gfx1250.
//   out[m][j] = sum_{i,k} B_k(x[m][i]) * kernel[k][i][j]
// GEMM view: M=16384, K=2048 (kk = i*8 + k), N=256.
//
// Block = 256 threads = 8 waves = 4 M-strips x 2 K-halves:
//   wave w: mt = w>>1 (16 output rows), kg = w&1 (kc = kg, kg+2, kg+4, ...).
// Each wave computes ALL 16 N-tiles for its K half, so one basis evaluation
// (A fragment) feeds 16 WMMAs -> VALU fully hidden under the matrix pipe.
// K-reduction: odd-K wave spills 16x(16x16) f32 partials to LDS, barrier,
// even-K wave adds and stores.
// ---------------------------------------------------------------------------

typedef __attribute__((ext_vector_type(16))) _Float16 v16h;
typedef __attribute__((ext_vector_type(8)))  float    v8f;
typedef __attribute__((ext_vector_type(4)))  float    v4f;

#define IN_CH    256
#define OUT_CH   256
#define BASIS    8
#define KDIM     (IN_CH * BASIS)       // 2048
#define MTILE    64
#define KC_STEPS (KDIM / 32)           // 64 K-chunks of 32
#define NTILES   (OUT_CH / 16)         // 16
// Wp: one 1KB "fragment" per (kc, nt): 32 lanes * 8 VGPRs * 2 f16 = 512 f16.
// fragment f = kc*16 + nt; byte addr = f*1024 + chunk*512 + lane*16.

// ---- cubic B-spline bases, uniform knots from make_grid() -----------------
// grid = linspace(-2.2, 2.2, 12), h = 0.4  ->  u = (x + 2.2)/0.4 = 2.5x + 5.5
// Truncated-power form of the cardinal cubic (== Cox-de Boor recursion):
//   B_t(x) = (1/6) [ p(u-t) - 4 p(u-t-1) + 6 p(u-t-2) - 4 p(u-t-3) + p(u-t-4) ]
//   p(y) = max(y,0)^3,  t = 0..7
__device__ __forceinline__ void bases8(float xv, float* o) {
  const float u = fmaf(xv, 2.5f, 5.5f);
  float p[12];
#pragma unroll
  for (int j = 0; j < 12; ++j) {
    float v = fmaxf(u - (float)j, 0.0f);
    p[j] = v * v * v;
  }
#pragma unroll
  for (int t = 0; t < 8; ++t) {
    float b = p[t] + p[t + 4] + 6.0f * p[t + 2] - 4.0f * (p[t + 1] + p[t + 3]);
    o[t] = b * 0.16666667f;
  }
}

// ---- W preprocessing: kernel[k][i][j] (f32) -> Wp in B-fragment layout ----
// 16-bit B-matrix 32x16 layout (ISA B tables):
//   lanes 0-15  : column N = lane,    VGPR r holds K = 2r, 2r+1 (r<4), 8+2r, 9+2r (r>=4)
//   lanes 16-31 : column N = lane-16, VGPR r holds K = 8+2r, 9+2r (r<4), 16+2r, 17+2r (r>=4)
// One thread writes one packed f16 pair (uint32).
__global__ __launch_bounds__(256) void kan_wprep(const float* __restrict__ sk,
                                                 uint32_t* __restrict__ Wp) {
  const int p = blockIdx.x * blockDim.x + threadIdx.x;   // 0 .. 262143 pairs
  const int f = p >> 8;            // fragment 0..1023
  const int q = p & 255;           // pair within fragment
  const int c = q >> 7;            // 16B chunk (VGPRs 0-3 vs 4-7)
  const int L = (q >> 2) & 31;     // lane
  const int t = q & 3;             // VGPR within chunk
  const int kc = f >> 4;
  const int nt = f & 15;
  const int r  = c * 4 + t;                               // VGPR 0..7
  const int kkbase = kc * 32 + ((L & 16) ? 8 : 0) + (c ? 8 : 0) + 2 * r;
  const int j = nt * 16 + (L & 15);

  int kk = kkbase;
  int i = kk >> 3, k = kk & 7;
  const float lo = sk[(k << 16) + (i << 8) + j];          // kernel[k][i][j]
  kk = kkbase + 1;
  i = kk >> 3; k = kk & 7;
  const float hi = sk[(k << 16) + (i << 8) + j];

  union { uint32_t u; _Float16 h[2]; } pk;
  pk.h[0] = (_Float16)lo;
  pk.h[1] = (_Float16)hi;
  Wp[p] = pk.u;
}

// ---- fused main kernel ----------------------------------------------------
// A fragment built in registers (16-bit A 16x32 layout):
//   lanes 0-15  row M=lane   : elems 0..7 -> K 0..7,  elems 8..15 -> K 16..23
//   lanes 16-31 row M=lane-16: elems 0..7 -> K 8..15, elems 8..15 -> K 24..31
// With kk = i*8+k, K-chunk kc = channels {4kc..4kc+3}; lane owns channels
// ia = 4kc + (lane>=16) and ia+2 -> two bases8() calls fill all 8 A VGPRs.
__global__ __launch_bounds__(256) void kan_main(const float* __restrict__ x,
                                                const uint32_t* __restrict__ Wp,
                                                float* __restrict__ out) {
  __shared__ float red[4][32][NTILES * 8];   // 64 KB: [mt][lane][tile*8 + r]

  const int lane = threadIdx.x & 31;
  const int wave = threadIdx.x >> 5;
  const int mt   = wave >> 1;     // M strip within block
  const int kg   = wave & 1;      // K half (even/odd kc)
  const int lrow = lane & 15;
  const int lhi  = lane >> 4;

  const int m = blockIdx.x * MTILE + mt * 16 + lrow;
  const float* __restrict__ xr = x + (size_t)m * IN_CH;

  v8f acc[NTILES];
#pragma unroll
  for (int t = 0; t < NTILES; ++t) acc[t] = (v8f)(0.0f);

  for (int kc = kg; kc < KC_STEPS; kc += 2) {
    // ---- build A fragment in registers ----
    const int ia = kc * 4 + lhi;
    const float xa = xr[ia];
    const float xb = xr[ia + 2];
    float ba[8], bb[8];
    bases8(xa, ba);
    bases8(xb, bb);
    v16h a;
#pragma unroll
    for (int e = 0; e < 8; ++e) {
      a[e]     = (_Float16)ba[e];
      a[e + 8] = (_Float16)bb[e];
    }

    // ---- all 16 N-tiles: pre-swizzled B fragments (L2-resident) + WMMA ----
#pragma unroll
    for (int t = 0; t < NTILES; ++t) {
      const int f = kc * 16 + t;
      const uint32_t* fb = Wp + f * 256 + lane * 4;
      union { struct { v4f lo, hi; } p; v16h h; } u;
      u.p.lo = *(const v4f*)(fb);         // VGPRs 0-3 (16B)
      u.p.hi = *(const v4f*)(fb + 128);   // VGPRs 4-7 (16B)
      acc[t] = __builtin_amdgcn_wmma_f32_16x16x32_f16(
          /*neg_a=*/false, a, /*neg_b=*/false, u.h,
          /*c_mod=*/(short)0, acc[t], /*reuse_a=*/false, /*reuse_b=*/false);
    }
  }

  // ---- K-reduction across the wave pair via LDS ----
  if (kg == 1) {
#pragma unroll
    for (int t = 0; t < NTILES; ++t)
      *(v8f*)&red[mt][lane][t * 8] = acc[t];     // 2x ds_store_b128 each
  }
  __syncthreads();

  if (kg == 0) {
    // C layout per 16x16 f32 tile: VGPR r, lanes 0-15 -> M=r,
    // lanes 16-31 -> M=r+8; N = lane&15.
    const int m0 = blockIdx.x * MTILE + mt * 16 + lhi * 8;
#pragma unroll
    for (int t = 0; t < NTILES; ++t) {
      const v8f part = *(const v8f*)&red[mt][lane][t * 8];
      const v8f c = acc[t] + part;
      const int col = t * 16 + lrow;
#pragma unroll
      for (int r = 0; r < 8; ++r) {
        out[(size_t)(m0 + r) * OUT_CH + col] = c[r];
      }
    }
  }
}

extern "C" void kernel_launch(void* const* d_in, const int* in_sizes, int n_in,
                              void* d_out, int out_size, void* d_ws, size_t ws_size,
                              hipStream_t stream) {
  const float* x  = (const float*)d_in[0];          // (4, 4096, 256) f32
  const float* sk = (const float*)d_in[1];          // (8, 256, 256) f32
  // d_in[2] = grid: deterministic linspace(-2.2, 2.2, 12); folded into bases8.
  float* out      = (float*)d_out;                  // (4, 4096, 256) f32
  uint32_t* Wp    = (uint32_t*)d_ws;                // 1 MB swizzled f16 weights

  // 262144 f16 pairs / 256 threads
  kan_wprep<<<dim3(1024), dim3(256), 0, stream>>>(sk, Wp);
  // 16384 tokens / 64 per block
  kan_main<<<dim3(256), dim3(256), 0, stream>>>(x, Wp, out);
}
